// MPNNNodeClassifier_60344290509247
// MI455X (gfx1250) — compile-verified, matrix operand order checked by reading, not run on hardware
//
#include <hip/hip_runtime.h>
#include <hip/hip_bf16.h>
#include <math.h>

// ---------------------------------------------------------------------------
// MPNN node classifier for MI455X (gfx1250), wave32 + WMMA (fp32, exact).
//
// Math: conv(x) = scatter_add_dst( relu(x[src] @ W + b) )  ==
//       y = relu(x @ W + b); h = y (self loop); h[dst] += y[src] per edge.
// relu after each conv is the identity (sum of non-negative messages).
// ---------------------------------------------------------------------------

typedef float v2f  __attribute__((ext_vector_type(2)));
typedef float v8f  __attribute__((ext_vector_type(8)));
typedef __attribute__((ext_vector_type(16))) _Float16 v16h;

#if defined(__has_builtin)
#  if __has_builtin(__builtin_amdgcn_wmma_f32_16x16x4_f32)
#    define USE_F32_WMMA 1
#  endif
#endif

// One wave computes a 64x16 strip of  Y = act(A @ B + bias): 4 M-subtiles
// sharing each B fragment (4x B reuse, 4 WMMAs per B load pair).
// A: [M,K] row-major, B: [K,Ncols] row-major. K % 32 == 0.
// M tail rows clamped on load, guarded on store. Ncols guarded by a 0/1
// multiplicative mask on B loads (branchless; no EXEC manipulation).
__global__ __launch_bounds__(32)
void gemm_bias_relu_wmma(const float* __restrict__ A,
                         const float* __restrict__ B,
                         const float* __restrict__ bias,
                         float* __restrict__ Y,
                         int M, int K, int Ncols, int do_relu)
{
    const int lane = threadIdx.x & 31;
    const int m0   = blockIdx.x * 64;
    const int n0   = blockIdx.y * 16;

    // Column handling (branchless): clamp + multiplicative mask.
    const int   colg = n0 + (lane & 15);
    const float msk  = (colg < Ncols) ? 1.0f : 0.0f;
    const int   colc = min(colg, Ncols - 1);

    // V_WMMA_F32_16X16X4_F32 fragment layout: lanes 0-15 hold K=koff{0,1},
    // lanes 16-31 hold K=koff{2,3}; lane&15 is the M-row / N-col.
    const int koff = (lane >> 4) * 2;

    const int r0 = min(m0 +      (lane & 15), M - 1);
    const int r1 = min(m0 + 16 + (lane & 15), M - 1);
    const int r2 = min(m0 + 32 + (lane & 15), M - 1);
    const int r3 = min(m0 + 48 + (lane & 15), M - 1);
    const float* __restrict__ A0 = A + (size_t)r0 * K + koff;
    const float* __restrict__ A1 = A + (size_t)r1 * K + koff;
    const float* __restrict__ A2 = A + (size_t)r2 * K + koff;
    const float* __restrict__ A3 = A + (size_t)r3 * K + koff;
    const float* __restrict__ Bp = B + (size_t)koff * Ncols + colc;

    v8f acc0 = {}, acc1 = {}, acc2 = {}, acc3 = {};

#if USE_F32_WMMA
    #pragma unroll 4
    for (int k = 0; k < K; k += 4) {
        v2f b;
        b.x = Bp[0]     * msk;
        b.y = Bp[Ncols] * msk;
        v2f a0 = *(const v2f*)(A0 + k);   // aligned pair (k%4==0, koff even)
        v2f a1 = *(const v2f*)(A1 + k);
        v2f a2 = *(const v2f*)(A2 + k);
        v2f a3 = *(const v2f*)(A3 + k);
        acc0 = __builtin_amdgcn_wmma_f32_16x16x4_f32(false, a0, false, b, (short)0, acc0, false, false);
        acc1 = __builtin_amdgcn_wmma_f32_16x16x4_f32(false, a1, false, b, (short)0, acc1, false, false);
        acc2 = __builtin_amdgcn_wmma_f32_16x16x4_f32(false, a2, false, b, (short)0, acc2, false, false);
        acc3 = __builtin_amdgcn_wmma_f32_16x16x4_f32(false, a3, false, b, (short)0, acc3, false, false);
        Bp += (size_t)4 * Ncols;
    }
#else
    // Fallback: f16 inputs / f32 accumulate (codegen-confirmed builtin).
    const int kbase = (lane >> 4) * 8;
    for (int k = 0; k < K; k += 32) {
        v16h a0, a1, a2, a3, b;
        #pragma unroll
        for (int j = 0; j < 16; ++j) {
            int kk = ((j >> 3) * 16) + kbase + (j & 7);     // K offset in step
            b[j]  = (_Float16)(B[(size_t)(k + kk) * Ncols + colc] * msk);
            a0[j] = (_Float16)A0[k + kk - koff];
            a1[j] = (_Float16)A1[k + kk - koff];
            a2[j] = (_Float16)A2[k + kk - koff];
            a3[j] = (_Float16)A3[k + kk - koff];
        }
        acc0 = __builtin_amdgcn_wmma_f32_16x16x32_f16(false, a0, false, b, (short)0, acc0, false, false);
        acc1 = __builtin_amdgcn_wmma_f32_16x16x32_f16(false, a1, false, b, (short)0, acc1, false, false);
        acc2 = __builtin_amdgcn_wmma_f32_16x16x32_f16(false, a2, false, b, (short)0, acc2, false, false);
        acc3 = __builtin_amdgcn_wmma_f32_16x16x32_f16(false, a3, false, b, (short)0, acc3, false, false);
    }
#endif

    // C/D layout: VGPR r, lanes 0-15 -> (M=r, N=lane); lanes 16-31 -> (M=8+r).
    if (colg < Ncols) {
        const int   mhi = (lane >> 4) * 8;
        const float bv  = bias[colc];
        v8f accs[4] = {acc0, acc1, acc2, acc3};
        #pragma unroll
        for (int s = 0; s < 4; ++s) {
            #pragma unroll
            for (int r = 0; r < 8; ++r) {
                int m = m0 + s * 16 + mhi + r;
                if (m < M) {
                    float v = accs[s][r] + bv;
                    if (do_relu) v = fmaxf(v, 0.0f);
                    Y[(size_t)m * Ncols + colg] = v;
                }
            }
        }
    }
}

// h[dst] += y[src], one wave per edge, lane-strided over H features.
__global__ __launch_bounds__(256)
void edge_aggregate(const int* __restrict__ src, const int* __restrict__ dst,
                    const float* __restrict__ y, float* __restrict__ h,
                    int E, int H)
{
    int e = blockIdx.x * (blockDim.x >> 5) + (threadIdx.x >> 5);
    if (e >= E) return;
    const int lane = threadIdx.x & 31;
    const float* __restrict__ ys = y + (size_t)src[e] * H;
    float* __restrict__       hd = h + (size_t)dst[e] * H;
    #pragma unroll 8
    for (int c = lane; c < H; c += 32)
        atomicAdd(&hd[c], ys[c]);
}

__global__ __launch_bounds__(256)
void copy_f32(const float* __restrict__ s, float* __restrict__ d, size_t n)
{
    size_t i      = (size_t)blockIdx.x * blockDim.x + threadIdx.x;
    size_t stride = (size_t)gridDim.x * blockDim.x;
    for (; i < n; i += stride) d[i] = s[i];
}

__global__ __launch_bounds__(256)
void zero_f32(float* __restrict__ d, size_t n)
{
    size_t i      = (size_t)blockIdx.x * blockDim.x + threadIdx.x;
    size_t stride = (size_t)gridDim.x * blockDim.x;
    for (; i < n; i += stride) d[i] = 0.0f;
}

// sc[row[i], :] += val[i] * lg[col[i], :]   (one 64-thread block per nnz)
__global__ __launch_bounds__(64)
void pvt_scatter(const int* __restrict__ row, const int* __restrict__ col,
                 const float* __restrict__ val,
                 const float* __restrict__ lg, float* __restrict__ sc, int C)
{
    const int i = blockIdx.x;
    const int t = threadIdx.x;
    if (t >= C) return;
    const float v = val[i];
    atomicAdd(&sc[(size_t)row[i] * C + t], v * lg[(size_t)col[i] * C + t]);
}

// Wave-per-row log_softmax, wave32 shuffle reductions (C <= 64).
__global__ __launch_bounds__(256)
void log_softmax_rows(const float* __restrict__ x, float* __restrict__ out,
                      int N, int C)
{
    int r = blockIdx.x * (blockDim.x >> 5) + (threadIdx.x >> 5);
    if (r >= N) return;
    const int lane = threadIdx.x & 31;
    const float* __restrict__ xr = x + (size_t)r * C;

    float v0 = (lane      < C) ? xr[lane]      : -INFINITY;
    float v1 = (lane + 32 < C) ? xr[lane + 32] : -INFINITY;

    float m = fmaxf(v0, v1);
    for (int off = 16; off; off >>= 1) m = fmaxf(m, __shfl_xor(m, off, 32));

    float s = 0.0f;
    if (lane      < C) s += expf(v0 - m);
    if (lane + 32 < C) s += expf(v1 - m);
    for (int off = 16; off; off >>= 1) s += __shfl_xor(s, off, 32);

    const float lse = m + logf(s);
    if (lane      < C) out[(size_t)r * C + lane]      = v0 - lse;
    if (lane + 32 < C) out[(size_t)r * C + lane + 32] = v1 - lse;
}

extern "C" void kernel_launch(void* const* d_in, const int* in_sizes, int n_in,
                              void* d_out, int out_size, void* d_ws, size_t ws_size,
                              hipStream_t stream)
{
    const float* x    = (const float*)d_in[0];
    const int*   esrc = (const int*)  d_in[1];
    const int*   edst = (const int*)  d_in[2];
    const int*   prow = (const int*)  d_in[3];
    const int*   pcol = (const int*)  d_in[4];
    const float* pval = (const float*)d_in[5];
    const float* w1   = (const float*)d_in[6];
    const float* b1   = (const float*)d_in[7];
    const float* w2   = (const float*)d_in[8];
    const float* b2   = (const float*)d_in[9];
    const float* wfc  = (const float*)d_in[10];
    const float* bfc  = (const float*)d_in[11];

    const int H   = in_sizes[7];            // 256
    const int C   = in_sizes[11];           // 40
    const int F   = in_sizes[6] / H;        // 256
    const int N   = in_sizes[0] / F;        // 50000
    const int E   = in_sizes[1];            // 1.6M
    const int NNZ = in_sizes[3];            // 50000

    // Workspace layout (102.4 MB): y | h ; logits & scatter alias dead y.
    float* y  = (float*)d_ws;
    float* h  = y + (size_t)N * H;
    float* lg = y;                          // y dead after conv2 aggregation
    float* sc = lg + (size_t)N * C;

    const int mt64 = (N + 63) / 64;
    const size_t nh = (size_t)N * H;

    // ---- conv1: y = relu(x @ w1 + b1); h = y + scatter(y[src] -> dst) ----
    gemm_bias_relu_wmma<<<dim3(mt64, H / 16), 32, 0, stream>>>(x, w1, b1, y, N, F, H, 1);
    copy_f32<<<4096, 256, 0, stream>>>(y, h, nh);
    edge_aggregate<<<(E + 7) / 8, 256, 0, stream>>>(esrc, edst, y, h, E, H);

    // ---- conv2 (ping-pong buffers) ----
    gemm_bias_relu_wmma<<<dim3(mt64, H / 16), 32, 0, stream>>>(h, w2, b2, y, N, H, H, 1);
    copy_f32<<<4096, 256, 0, stream>>>(y, h, nh);
    edge_aggregate<<<(E + 7) / 8, 256, 0, stream>>>(esrc, edst, y, h, E, H);

    // ---- FC head: lg = h @ wfc + bfc ----
    gemm_bias_relu_wmma<<<dim3(mt64, (C + 15) / 16), 32, 0, stream>>>(h, wfc, bfc, lg, N, H, C, 0);

    // ---- sparse PvT @ lg ----
    zero_f32<<<2048, 256, 0, stream>>>(sc, (size_t)N * C);
    pvt_scatter<<<NNZ, 64, 0, stream>>>(prow, pcol, pval, lg, sc, C);

    // ---- log_softmax -> d_out ----
    log_softmax_rows<<<(N + 7) / 8, 256, 0, stream>>>(sc, (float*)d_out, N, C);
}
